// GroupedQueryAttention_18038862643563
// MI455X (gfx1250) — matrix operand order repaired and checked by reading, not compiled, not run on hardware
//
#include <hip/hip_runtime.h>
#include <hip/hip_bf16.h>
#include <math.h>

typedef _Float16 f16;
typedef __attribute__((ext_vector_type(16))) _Float16 v16h;
typedef __attribute__((ext_vector_type(8)))  _Float16 v8h;
typedef __attribute__((ext_vector_type(8)))  float    v8f;

#define S_LEN 2048
#define E_DIM 4096
#define HQ_N  32
#define HK_N  8
#define D_HD  128

// ---------------------------------------------------------------------------
// WMMA helpers (CDNA5 16x16x32 f16 -> f32, VGPR layouts per CDNA5 ISA)
// ---------------------------------------------------------------------------
__device__ __forceinline__ v8f wmma16(v16h a, v16h b, v8f c) {
  // 8 args: (neg_a, A, neg_b, B, c_mod, C, reuse_a, reuse_b)
  return __builtin_amdgcn_wmma_f32_16x16x32_f16(false, a, false, b, (short)0, c,
                                                false, false);
}

// A fragment: 16(M) x 32(K) f16 tile, row-major at `A` with leading dim ld.
// Lane L holds row M=L&15; j=0..7 -> K = (L>>4)*8 + j ; j=8..15 -> +16.
__device__ __forceinline__ v16h load_a_frag(const f16* A, int ld, int lane) {
  const int half = lane >> 4, r = lane & 15;
  const f16* p = A + r * ld + half * 8;
  v8h lo = *(const v8h*)(p);
  v8h hi = *(const v8h*)(p + 16);
  v16h out;
#pragma unroll
  for (int j = 0; j < 8; ++j) { out[j] = lo[j]; out[j + 8] = hi[j]; }
  return out;
}

// B fragment: 32(K) x 16(N). Memory holds it "transposed": element (k,n) at
// W[n*ld + k] (i.e. a row-major N x K panel). Lane L holds column N=L&15,
// j=0..7 -> K = (L>>4)*8 + j ; j=8..15 -> +16 (same as A fragment).
__device__ __forceinline__ v16h load_b_frag(const f16* W, int ld, int lane) {
  const int half = lane >> 4, n = lane & 15;
  const f16* p = W + n * ld + half * 8;
  v8h lo = *(const v8h*)(p);
  v8h hi = *(const v8h*)(p + 16);
  v16h out;
#pragma unroll
  for (int j = 0; j < 8; ++j) { out[j] = lo[j]; out[j + 8] = hi[j]; }
  return out;
}

// ---------------------------------------------------------------------------
// f32 -> f16 conversion
// ---------------------------------------------------------------------------
__global__ __launch_bounds__(256) void cvt_kernel(const float* __restrict__ s,
                                                  f16* __restrict__ d, long n) {
  long i = ((long)blockIdx.x * blockDim.x + threadIdx.x) * 4;
  if (i >= n) return;
  float4 v = *(const float4*)(s + i);
  d[i + 0] = (f16)v.x; d[i + 1] = (f16)v.y;
  d[i + 2] = (f16)v.z; d[i + 3] = (f16)v.w;
}

// ---------------------------------------------------------------------------
// C[M,N] = A[M,K] * W[N,K]^T  (f16 in, OutT out, f32 accumulate)
// block = 256 threads (8 waves, 4x2), block tile 128x64, K chunk 32 via LDS
// ---------------------------------------------------------------------------
template <typename OutT>
__global__ __launch_bounds__(256) void gemm_xwT(OutT* __restrict__ C,
                                                const f16* __restrict__ A,
                                                const f16* __restrict__ W,
                                                int N, int K) {
  __shared__ f16 As[128][40];  // padded: 40 f16 = 80B rows, 16B aligned
  __shared__ f16 Ws[64][40];
  const int tid = threadIdx.x;
  const int wave = tid >> 5, lane = tid & 31;
  const int wr = wave >> 1, wc = wave & 1;  // 4 x 2 wave grid
  const long m0 = (long)blockIdx.y * 128;
  const long n0 = (long)blockIdx.x * 64;
  v8f acc[2][2] = {};
  for (int k0 = 0; k0 < K; k0 += 32) {
    __syncthreads();
    {
      const int row = tid >> 1, seg = tid & 1;  // 128 rows x two 16-f16 halves
      const f16* g = A + (m0 + row) * (long)K + k0 + seg * 16;
      v8h a0 = *(const v8h*)g;
      v8h a1 = *(const v8h*)(g + 8);
      *(v8h*)&As[row][seg * 16]     = a0;
      *(v8h*)&As[row][seg * 16 + 8] = a1;
    }
    if (tid < 128) {
      const int rw = tid >> 1, sg = tid & 1;    // 64 rows x two halves
      const f16* gw = W + (n0 + rw) * (long)K + k0 + sg * 16;
      v8h b0 = *(const v8h*)gw;
      v8h b1 = *(const v8h*)(gw + 8);
      *(v8h*)&Ws[rw][sg * 16]     = b0;
      *(v8h*)&Ws[rw][sg * 16 + 8] = b1;
    }
    __syncthreads();
    v16h af[2], bf[2];
#pragma unroll
    for (int i = 0; i < 2; ++i) af[i] = load_a_frag(&As[wr * 32 + i * 16][0], 40, lane);
#pragma unroll
    for (int j = 0; j < 2; ++j) bf[j] = load_b_frag(&Ws[wc * 32 + j * 16][0], 40, lane);
#pragma unroll
    for (int i = 0; i < 2; ++i)
#pragma unroll
      for (int j = 0; j < 2; ++j) acc[i][j] = wmma16(af[i], bf[j], acc[i][j]);
  }
  const int half = lane >> 4, ln = lane & 15;
#pragma unroll
  for (int i = 0; i < 2; ++i)
#pragma unroll
    for (int j = 0; j < 2; ++j) {
      OutT* p = C + (m0 + wr * 32 + i * 16 + half * 8) * (long)N +
                n0 + wc * 32 + j * 16 + ln;
#pragma unroll
      for (int r = 0; r < 8; ++r) p[(long)r * N] = (OutT)acc[i][j][r];
    }
}

// ---------------------------------------------------------------------------
// In-place RoPE on f16 tensor laid out [S][H][128]; one thread per (s,h,pair)
// ---------------------------------------------------------------------------
__global__ __launch_bounds__(256) void rope_kernel(f16* __restrict__ t, int H,
                                                   long total) {
  long idx = (long)blockIdx.x * blockDim.x + threadIdx.x;
  if (idx >= total) return;
  const int pair = (int)(idx & 63);
  long sh = idx >> 6;
  const int h = (int)(sh % H);
  const long s = sh / H;
  // inv_freq = 10000^(-pair/64) = 2^(-pair*log2(10000)/64)
  const float inv = exp2f(-(float)pair * (13.287712379549449f / 64.0f));
  const float th = (float)s * inv;
  float sn, cs;
  __sincosf(th, &sn, &cs);
  f16* p = t + (s * (long)H + h) * D_HD + pair;
  const float x1 = (float)p[0];
  const float x2 = (float)p[64];
  p[0]  = (f16)(x1 * cs - x2 * sn);
  p[64] = (f16)(x2 * cs + x1 * sn);
}

// ---------------------------------------------------------------------------
// Fused causal GQA flash attention.
// grid = (S/64, HQ); block = 128 threads = 4 waves; each wave: 16 queries,
// full D=128 accumulator in registers; key blocks of 32 staged in LDS.
// ---------------------------------------------------------------------------
__global__ __launch_bounds__(128) void attn_kernel(const f16* __restrict__ Q,
                                                   const f16* __restrict__ Kc,
                                                   const f16* __restrict__ Vc,
                                                   f16* __restrict__ AO) {
  __shared__ f16 Kt[32][136];     // [key][d], padded
  __shared__ f16 Vt[128][40];     // transposed: [d][key], padded
  __shared__ f16 Pt[4][16][40];   // per-wave probs tile [q][key(0..31)]
  const int h = blockIdx.y;
  const int hk = h >> 2;          // HQ/HK = 4
  const int tid = threadIdx.x;
  const int wave = tid >> 5, lane = tid & 31;
  const int half = lane >> 4, ln = lane & 15;
  const long qblock = (long)blockIdx.x * 64;
  const long q0 = qblock + wave * 16;

  // Preload the wave's Q fragments (16 x 128 = 4 k-chunks of 32)
  v16h qf[4];
  const f16* Qp = Q + q0 * (long)(HQ_N * D_HD) + h * D_HD;
#pragma unroll
  for (int c = 0; c < 4; ++c) qf[c] = load_a_frag(Qp + c * 32, HQ_N * D_HD, lane);

  v8f acc[8] = {};                // 8 d-tiles of 16x16
  float mrow[8], lrow[8];
#pragma unroll
  for (int r = 0; r < 8; ++r) { mrow[r] = -3.0e38f; lrow[r] = 0.0f; }

  // softmax in base-2 domain: s * (1/sqrt(128)) * log2(e)
  const float sm_scale = 0.08838834764831845f * 1.4426950408889634f;
  const int nkb = (int)(qblock >> 5) + 2;  // key blocks of 32 up to diagonal
  const long myqhi = q0 + 15;

  for (int kb = 0; kb < nkb; ++kb) {
    const long k0 = (long)kb * 32;
    __syncthreads();
    {
      // stage K (32x128) and V (transposed) — 128 threads: row=tid>>2, seg=tid&3
      const int row = tid >> 2, seg = tid & 3;
      const f16* gk = Kc + (k0 + row) * (long)(HK_N * D_HD) + hk * D_HD + seg * 32;
      v8h a0 = *(const v8h*)gk;
      v8h a1 = *(const v8h*)(gk + 8);
      v8h a2 = *(const v8h*)(gk + 16);
      v8h a3 = *(const v8h*)(gk + 24);
      f16* dk = &Kt[row][seg * 32];
      *(v8h*)dk = a0; *(v8h*)(dk + 8) = a1;
      *(v8h*)(dk + 16) = a2; *(v8h*)(dk + 24) = a3;
      const f16* gv = Vc + (k0 + row) * (long)(HK_N * D_HD) + hk * D_HD + seg * 32;
      v8h b0 = *(const v8h*)gv;
      v8h b1 = *(const v8h*)(gv + 8);
      v8h b2 = *(const v8h*)(gv + 16);
      v8h b3 = *(const v8h*)(gv + 24);
#pragma unroll
      for (int e = 0; e < 8; ++e) {
        Vt[seg * 32 + e][row]      = b0[e];
        Vt[seg * 32 + 8 + e][row]  = b1[e];
        Vt[seg * 32 + 16 + e][row] = b2[e];
        Vt[seg * 32 + 24 + e][row] = b3[e];
      }
    }
    __syncthreads();
    if (k0 <= myqhi) {   // wave-uniform
      // ---- scores: 16 queries x 32 keys, contraction over D=128 ----
      v8f sc0 = {}, sc1 = {};
#pragma unroll
      for (int c = 0; c < 4; ++c) {
        v16h kf0 = load_b_frag(&Kt[0][c * 32], 136, lane);
        v16h kf1 = load_b_frag(&Kt[16][c * 32], 136, lane);
        sc0 = wmma16(qf[c], kf0, sc0);
        sc1 = wmma16(qf[c], kf1, sc1);
      }
      // ---- causal mask + online softmax (rows striped per C layout) ----
      float p0a[8], p1a[8];
#pragma unroll
      for (int r = 0; r < 8; ++r) {
        const int qg = (int)q0 + half * 8 + r;
        const int kg0 = (int)k0 + ln;
        float s0 = (kg0 <= qg)      ? sc0[r] * sm_scale : -3.0e38f;
        float s1 = (kg0 + 16 <= qg) ? sc1[r] * sm_scale : -3.0e38f;
        float x = fmaxf(s0, s1);
        x = fmaxf(x, __shfl_xor(x, 1, 32));
        x = fmaxf(x, __shfl_xor(x, 2, 32));
        x = fmaxf(x, __shfl_xor(x, 4, 32));
        x = fmaxf(x, __shfl_xor(x, 8, 32));
        const float mn = fmaxf(mrow[r], x);
        const float esc = exp2f(mrow[r] - mn);
        mrow[r] = mn;
        const float p0 = exp2f(s0 - mn);
        const float p1 = exp2f(s1 - mn);
        float rs = p0 + p1;
        rs += __shfl_xor(rs, 1, 32);
        rs += __shfl_xor(rs, 2, 32);
        rs += __shfl_xor(rs, 4, 32);
        rs += __shfl_xor(rs, 8, 32);
        lrow[r] = lrow[r] * esc + rs;
#pragma unroll
        for (int t = 0; t < 8; ++t) acc[t][r] *= esc;
        p0a[r] = p0; p1a[r] = p1;
      }
      // ---- P relayout (C-layout -> A-layout) through per-wave LDS tile ----
#pragma unroll
      for (int r = 0; r < 8; ++r) {
        const int row = half * 8 + r;
        Pt[wave][row][ln]      = (f16)p0a[r];
        Pt[wave][row][16 + ln] = (f16)p1a[r];
      }
      asm volatile("s_wait_dscnt 0" ::: "memory");  // wave-local LDS RAW
      v16h pf = load_a_frag(&Pt[wave][0][0], 40, lane);
      // ---- O += P * V  (8 d-tiles, contraction over 32 keys) ----
#pragma unroll
      for (int t = 0; t < 8; ++t) {
        v16h vf = load_b_frag(&Vt[t * 16][0], 40, lane);
        acc[t] = wmma16(pf, vf, acc[t]);
      }
    }
  }
  // ---- normalize and store f16 attention output [S][HQ*D] ----
#pragma unroll
  for (int t = 0; t < 8; ++t) {
    f16* op = AO + (q0 + half * 8) * (long)(HQ_N * D_HD) + h * D_HD + t * 16 + ln;
#pragma unroll
    for (int r = 0; r < 8; ++r)
      op[(long)r * (HQ_N * D_HD)] = (f16)(acc[t][r] / lrow[r]);
  }
}

// ---------------------------------------------------------------------------
extern "C" void kernel_launch(void* const* d_in, const int* in_sizes, int n_in,
                              void* d_out, int out_size, void* d_ws, size_t ws_size,
                              hipStream_t stream) {
  (void)in_sizes; (void)n_in; (void)out_size; (void)ws_size;
  const float* x  = (const float*)d_in[0];
  const float* wq = (const float*)d_in[1];
  const float* wk = (const float*)d_in[2];
  const float* wv = (const float*)d_in[3];
  const float* wo = (const float*)d_in[4];
  float* out = (float*)d_out;
  f16* ws = (f16*)d_ws;

  const long SE = (long)S_LEN * E_DIM;          // 8M  : x / Q / AO
  const long EE = (long)E_DIM * E_DIM;          // 16M : wq / wo
  const long KE = (long)(HK_N * D_HD) * E_DIM;  // 4M  : wk / wv
  const long KV = (long)S_LEN * HK_N * D_HD;    // 2M  : K / V

  long o = 0;
  f16* x16  = ws + o; o += SE;
  f16* wq16 = ws + o; o += EE;
  f16* wk16 = ws + o; o += KE;
  f16* wv16 = ws + o; o += KE;
  f16* wo16 = ws + o; o += EE;
  f16* Qb   = ws + o; o += SE;
  f16* Kb   = ws + o; o += KV;
  f16* Vb   = ws + o; o += KV;
  f16* AO   = ws + o; o += SE;

  auto cvt = [&](const float* s, f16* d, long n) {
    unsigned blocks = (unsigned)((n / 4 + 255) / 256);
    cvt_kernel<<<dim3(blocks), dim3(256), 0, stream>>>(s, d, n);
  };
  cvt(x,  x16,  SE);
  cvt(wq, wq16, EE);
  cvt(wk, wk16, KE);
  cvt(wv, wv16, KE);
  cvt(wo, wo16, EE);

  // Q/K/V projections (f16 out)
  gemm_xwT<f16><<<dim3(E_DIM / 64, S_LEN / 128), 256, 0, stream>>>(
      Qb, x16, wq16, E_DIM, E_DIM);
  gemm_xwT<f16><<<dim3((HK_N * D_HD) / 64, S_LEN / 128), 256, 0, stream>>>(
      Kb, x16, wk16, HK_N * D_HD, E_DIM);
  gemm_xwT<f16><<<dim3((HK_N * D_HD) / 64, S_LEN / 128), 256, 0, stream>>>(
      Vb, x16, wv16, HK_N * D_HD, E_DIM);

  // RoPE on Q and K
  const long qtot = (long)S_LEN * HQ_N * 64;
  rope_kernel<<<dim3((unsigned)((qtot + 255) / 256)), 256, 0, stream>>>(Qb, HQ_N, qtot);
  const long ktot = (long)S_LEN * HK_N * 64;
  rope_kernel<<<dim3((unsigned)((ktot + 255) / 256)), 256, 0, stream>>>(Kb, HK_N, ktot);

  // Fused causal GQA attention
  attn_kernel<<<dim3(S_LEN / 64, HQ_N), 128, 0, stream>>>(Qb, Kb, Vb, AO);

  // Output projection (f32 out)
  gemm_xwT<float><<<dim3(E_DIM / 64, S_LEN / 128), 256, 0, stream>>>(
      out, AO, wo16, E_DIM, E_DIM);
}
